// ImageFeaturesMap_87771951661165
// MI455X (gfx1250) — matrix operand order repaired and checked by reading, not compiled, not run on hardware
//
#include <hip/hip_runtime.h>

// Problem constants (match reference)
constexpr int Bsz  = 8;
constexpr int Npts = 16384;
constexpr int Fdim = 128;
constexpr int Pdim = 28;
constexpr int Kdim = Pdim * Pdim;   // 784
constexpr int KP   = 800;           // K padded to multiple of 32 (25 chunks)
constexpr int KLS  = 808;           // LDS row stride (halfs), 16B-aligned
constexpr int KC   = KP / 32;       // 25 WMMA K-chunks
constexpr int MT   = 64;            // rows (points) per workgroup

typedef _Float16 v16h __attribute__((ext_vector_type(16)));
typedef _Float16 v8h  __attribute__((ext_vector_type(8)));
typedef _Float16 v2h  __attribute__((ext_vector_type(2)));
typedef float    v8f  __attribute__((ext_vector_type(8)));

union HV { v16h v; v8h h[2]; };

// Dynamic-LDS layout (bytes)
constexpr int SM_A   = 0;                         // A tile: MT x KLS halfs
constexpr int SM_K1  = SM_A  + MT * KLS * 2;      // k1h:    MT x 32 halfs
constexpr int SM_K0  = SM_K1 + MT * 32 * 2;       // k0h:    MT x 28 halfs
constexpr int SM_XS  = SM_K0 + MT * 28 * 2;       // xs:     MT x 2 floats
constexpr int SMEM_BYTES = SM_XS + MT * 2 * 4;    // ~112 KB (320 KB/WGP available)

// ---------------------------------------------------------------------------
// Prep: image_features [F, P, P] f32  ->  W [F, KP] f16 (zero-padded K tail).
// W[f][k] is exactly B[k][f] of the GEMM (column f contiguous in k).
// ---------------------------------------------------------------------------
__global__ __launch_bounds__(256)
void ImageFeaturesMap_prep_kernel(const float* __restrict__ IF,
                                  _Float16* __restrict__ W) {
    int idx = blockIdx.x * 256 + threadIdx.x;
    if (idx >= Fdim * KP) return;
    int f = idx / KP;
    int k = idx - f * KP;
    float v = (k < Kdim) ? IF[f * Kdim + k] : 0.0f;
    W[idx] = (_Float16)v;
}

// ---------------------------------------------------------------------------
// Main: out[m, f] = sum_k A[m,k] * W_t[k,f]
//   A[m, p*28+q] = exp(-inv*(g[p]-x0)^2) * exp(-inv*(g[q]-x1)^2)
// 256 threads = 8 waves. Wave w -> F-tile [16w, 16w+16); 4 M-subtiles of 16.
// ---------------------------------------------------------------------------
__global__ __launch_bounds__(256)
void ImageFeaturesMap_gemm_kernel(const float* __restrict__ x,
                                  const float* __restrict__ sigma,
                                  const _Float16* __restrict__ W,
                                  float* __restrict__ out) {
    extern __shared__ char smem[];
    _Float16* A   = (_Float16*)(smem + SM_A);    // [MT][KLS]
    _Float16* k1h = (_Float16*)(smem + SM_K1);   // [MT][32]
    _Float16* k0h = (_Float16*)(smem + SM_K0);   // [MT][28]
    float*    xs  = (float*)   (smem + SM_XS);   // [MT*2]

    const int tid   = threadIdx.x;
    const int mbase = blockIdx.x * MT;

    // stage x coords for this WG's rows
    if (tid < MT * 2) xs[tid] = x[mbase * 2 + tid];
    __syncthreads();

    const float ls  = __expf(sigma[0]);
    const float inv = 0.5f / (ls * ls);

    // Gaussian kernel values -> f16 (MT*Pdim = 1792 entries, 7/thread)
    for (int idx = tid; idx < MT * Pdim; idx += 256) {
        int m = idx / Pdim;
        int p = idx - m * Pdim;
        float g  = 0.001f + (float)p * (0.998f / 27.0f);
        float d0 = g - xs[m * 2 + 0];
        float d1 = g - xs[m * 2 + 1];
        k0h[m * Pdim + p] = (_Float16)__expf(-inv * d0 * d0);
        k1h[m * 32   + p] = (_Float16)__expf(-inv * d1 * d1);
    }
    __syncthreads();

    // Build f16 A-tile with packed math: 4 threads per row, 7 p's per thread.
    // A[m][p*28 + q] = k0h[m][p] * k1h[m][q]  (v_pk_mul_f16, dword LDS stores)
    {
        const int m     = tid >> 2;        // row 0..63
        const int pbase = tid & 3;         // p = pbase + 4*j, j = 0..6
        _Float16* Arow = A + m * KLS;

        v2h k1r[14];
#pragma unroll
        for (int q2 = 0; q2 < 14; ++q2)
            k1r[q2] = *(const v2h*)&k1h[m * 32 + 2 * q2];

#pragma unroll
        for (int j = 0; j < 7; ++j) {
            const int p = pbase + 4 * j;
            const _Float16 a = k0h[m * Pdim + p];
            v2h a2 = {a, a};
#pragma unroll
            for (int q2 = 0; q2 < 14; ++q2) {
                v2h r = a2 * k1r[q2];
                *(v2h*)&Arow[p * Pdim + 2 * q2] = r;
            }
        }
        // zero-pad k in [784, 800): 64 rows x 8 dwords = 512 dwords, 2/thread
        for (int z = tid; z < MT * 8; z += 256) {
            int zm = z >> 3;
            int zk = Kdim + 2 * (z & 7);
            *(v2h*)&A[zm * KLS + zk] = (v2h){(_Float16)0.0f, (_Float16)0.0f};
        }
    }
    __syncthreads();

    const int wave  = tid >> 5;
    const int lane  = tid & 31;
    const int lo    = lane & 15;       // row (A) / col (B) within tile
    const int hi    = lane >> 4;       // half-wave select
    const int fbase = wave * 16;

    v8f acc0 = {}, acc1 = {}, acc2 = {}, acc3 = {};

    // Per-subtile A row base pointers: all ds_load offsets become small
    // immediates (kb*2 + {0,16,32,48} bytes <= 1632), no per-iter address VALU.
    const _Float16* __restrict__ ar0 = A + (lo)      * KLS + hi * 8;
    const _Float16* __restrict__ ar1 = A + (16 + lo) * KLS + hi * 8;
    const _Float16* __restrict__ ar2 = A + (32 + lo) * KLS + hi * 8;
    const _Float16* __restrict__ ar3 = A + (48 + lo) * KLS + hi * 8;

    // B column pointer for this lane's F column (contiguous in k, L2-resident)
    const _Float16* __restrict__ bcol = W + (size_t)(fbase + lo) * KP + hi * 8;

#pragma unroll
    for (int kc = 0; kc < KC; ++kc) {
        const int kb = kc * 32;

        // B fragment (32x16): K = g*16 + hi*8 + off (matches A)
        HV b;
        b.h[0] = *(const v8h*)(bcol + kb);
        b.h[1] = *(const v8h*)(bcol + kb + 16);

        // A fragments (16x32 each): ISA layout K = g*16 + hi*8 + off
        HV a0, a1, a2, a3;
        a0.h[0] = *(const v8h*)(ar0 + kb);
        a0.h[1] = *(const v8h*)(ar0 + kb + 16);
        a1.h[0] = *(const v8h*)(ar1 + kb);
        a1.h[1] = *(const v8h*)(ar1 + kb + 16);
        a2.h[0] = *(const v8h*)(ar2 + kb);
        a2.h[1] = *(const v8h*)(ar2 + kb + 16);
        a3.h[0] = *(const v8h*)(ar3 + kb);
        a3.h[1] = *(const v8h*)(ar3 + kb + 16);

        acc0 = __builtin_amdgcn_wmma_f32_16x16x32_f16(
            false, a0.v, false, b.v, (short)0, acc0, false, false);
        acc1 = __builtin_amdgcn_wmma_f32_16x16x32_f16(
            false, a1.v, false, b.v, (short)0, acc1, false, false);
        acc2 = __builtin_amdgcn_wmma_f32_16x16x32_f16(
            false, a2.v, false, b.v, (short)0, acc2, false, false);
        acc3 = __builtin_amdgcn_wmma_f32_16x16x32_f16(
            false, a3.v, false, b.v, (short)0, acc3, false, false);
    }

    // D layout: VGPR r holds M = r (lanes 0-15) / M = 8+r (lanes 16-31), N = lo
    const int col = fbase + lo;
#pragma unroll
    for (int r = 0; r < 8; ++r) {
        out[(size_t)(mbase +      hi * 8 + r) * Fdim + col] = acc0[r];
        out[(size_t)(mbase + 16 + hi * 8 + r) * Fdim + col] = acc1[r];
        out[(size_t)(mbase + 32 + hi * 8 + r) * Fdim + col] = acc2[r];
        out[(size_t)(mbase + 48 + hi * 8 + r) * Fdim + col] = acc3[r];
    }
}

// ---------------------------------------------------------------------------
extern "C" void kernel_launch(void* const* d_in, const int* in_sizes, int n_in,
                              void* d_out, int out_size, void* d_ws, size_t ws_size,
                              hipStream_t stream) {
    const float* x     = (const float*)d_in[0];   // [B, N, 2] f32
    const float* sigma = (const float*)d_in[1];   // [1] f32
    const float* IF    = (const float*)d_in[2];   // [F, P, P] f32
    float*       out   = (float*)d_out;           // [B, N, F] f32
    _Float16*    W     = (_Float16*)d_ws;         // [F, KP] f16 (200 KB)

    const int rows = in_sizes[0] / 2;             // B*N = 131072

    // 1) convert/transpose-view image features to padded f16 weight matrix
    ImageFeaturesMap_prep_kernel<<<(Fdim * KP + 255) / 256, 256, 0, stream>>>(IF, W);

    // 2) fused kernel-eval + packed outer-product A-build + WMMA GEMM
    ImageFeaturesMap_gemm_kernel<<<rows / MT, 256, SMEM_BYTES, stream>>>(x, sigma, W, out);
}
